// SubGraphLayer_64269890617596
// MI455X (gfx1250) — compile-verified, matrix operand order checked by reading, not run on hardware
//
#include <hip/hip_runtime.h>

typedef __attribute__((ext_vector_type(16))) _Float16 v16h;
typedef __attribute__((ext_vector_type(8)))  float    v8f;
typedef __attribute__((ext_vector_type(4)))  float    v4f;
typedef __attribute__((ext_vector_type(4)))  int      v4i;

#define NEG_SLOPE 0.01f
#define LDS_STRIDE 68   // dwords; rows stay 16B-aligned, banks spread between half-waves

// Native CDNA5 float max atomic, non-returning, device scope, GVS addressing:
// 64-bit uniform base in SGPRs + 32-bit per-lane byte offset + literal IOFFSET.
#define ATOMIC_FMAX_OFF(boff, x, OFFLIT)                                          \
    asm volatile("global_atomic_max_num_f32 %0, %1, %2 offset:" #OFFLIT           \
                 " scope:SCOPE_DEV" :: "v"(boff), "v"(x), "s"(aggp))

__global__ void init_agg(float* __restrict__ agg, long n) {
    long i = (long)blockIdx.x * blockDim.x + threadIdx.x;
    if (i < n) agg[i] = __int_as_float(0xFF800000);   // -inf sentinel (empty segment)
}

__global__ __launch_bounds__(256) void fused_gemm_scatter(
    const float* __restrict__ feat, const float* __restrict__ W,
    const float* __restrict__ bias, const int* __restrict__ idx,
    float* __restrict__ out, float* __restrict__ aggp,
    int ntiles)
{
    __shared__ float lds_h[8][16 * LDS_STRIDE];   // per-wave h-tile staging (34 KB)

    const int lane = threadIdx.x & 31;
    const int wave = threadIdx.x >> 5;
    const int half = lane >> 4;        // which 16-lane group
    const int nidx = lane & 15;        // row (A) / col (B,D) index within tile

    // ---- Preload B = W^T once per wave (ISA 16-bit 16x32 operand layout) ----
    // lanes 0-15: K = k0+{0..7} & k0+{16..23}; lanes 16-31: K = k0+{8..15} & k0+{24..31}
    v16h bop[4][2];
#pragma unroll
    for (int nt = 0; nt < 4; ++nt) {
        const float* wrow = W + (long)(nt * 16 + nidx) * 64;
#pragma unroll
        for (int ks = 0; ks < 2; ++ks) {
            const v4f* wp = (const v4f*)(wrow + ks * 32 + half * 8);
            v4f w0 = wp[0], w1 = wp[1];       // K = k0..k0+7
            v4f w2 = wp[4], w3 = wp[5];       // K = k0+16..k0+23
#pragma unroll
            for (int j = 0; j < 4; ++j) {
                bop[nt][ks][j]      = (_Float16)w0[j];
                bop[nt][ks][j + 4]  = (_Float16)w1[j];
                bop[nt][ks][j + 8]  = (_Float16)w2[j];
                bop[nt][ks][j + 12] = (_Float16)w3[j];
            }
        }
    }
    const float bv[4] = { bias[nidx], bias[16 + nidx], bias[32 + nidx], bias[48 + nidx] };

    const int wave_gid = blockIdx.x * 8 + wave;
    const int wave_cnt = gridDim.x * 8;
    float* lds_wave = &lds_h[wave][0];
    float* lds_wr = lds_wave + (half * 8) * LDS_STRIDE + nidx;        // store base
    const float* lds_rd = lds_wave + half * LDS_STRIDE + nidx * 4;    // load base
    const int lane_boff = nidx * 4;    // byte offset of this lane's column

    for (int tile = wave_gid; tile < ntiles; tile += wave_cnt) {      // wave-uniform
        const long row0 = (long)tile * 16;

        // ---- A operands: 16 rows x 64 K, streamed non-temporally ----
        v16h aop[2];
        {
            const float* fr = feat + (row0 + nidx) * 64;
#pragma unroll
            for (int ks = 0; ks < 2; ++ks) {
                const v4f* fp = (const v4f*)(fr + ks * 32 + half * 8);
                v4f f0 = __builtin_nontemporal_load(fp + 0);
                v4f f1 = __builtin_nontemporal_load(fp + 1);
                v4f f2 = __builtin_nontemporal_load(fp + 4);
                v4f f3 = __builtin_nontemporal_load(fp + 5);
#pragma unroll
                for (int j = 0; j < 4; ++j) {
                    aop[ks][j]      = (_Float16)f0[j];
                    aop[ks][j + 4]  = (_Float16)f1[j];
                    aop[ks][j + 8]  = (_Float16)f2[j];
                    aop[ks][j + 12] = (_Float16)f3[j];
                }
            }
        }

        // ---- 32-bit atomic byte-offsets for this lane's 8 rows (GVS addressing) ----
        int boff[8];
        {
            const v4i* ip = (const v4i*)(idx + row0 + half * 8);
            v4i s0 = ip[0], s1 = ip[1];
#pragma unroll
            for (int v = 0; v < 4; ++v) {
                boff[v]     = (s0[v] << 8) + lane_boff;   // seg*64 floats + col
                boff[v + 4] = (s1[v] << 8) + lane_boff;
            }
        }

        // ---- 4 channel-tiles of WMMA: h = X*W^T + b ----
        v8f accs[4];
#pragma unroll
        for (int nt = 0; nt < 4; ++nt) {
            v8f acc;
#pragma unroll
            for (int v = 0; v < 8; ++v) acc[v] = bv[nt];   // bias is column-only
            acc = __builtin_amdgcn_wmma_f32_16x16x32_f16(
                      false, aop[0], false, bop[nt][0], (short)0, acc, false, false);
            accs[nt] = __builtin_amdgcn_wmma_f32_16x16x32_f16(
                      false, aop[1], false, bop[nt][1], (short)0, acc, false, false);
        }

        // ---- leaky_relu + LDS stage + scatter-max (immediate per-tile offsets) ----
#pragma unroll
        for (int v = 0; v < 8; ++v) {
            // leaky_relu(x) == max(x, slope*x) for 0 < slope < 1
            float x0 = fmaxf(accs[0][v], NEG_SLOPE * accs[0][v]);
            float x1 = fmaxf(accs[1][v], NEG_SLOPE * accs[1][v]);
            float x2 = fmaxf(accs[2][v], NEG_SLOPE * accs[2][v]);
            float x3 = fmaxf(accs[3][v], NEG_SLOPE * accs[3][v]);
            lds_wr[v * LDS_STRIDE +  0] = x0;
            lds_wr[v * LDS_STRIDE + 16] = x1;
            lds_wr[v * LDS_STRIDE + 32] = x2;
            lds_wr[v * LDS_STRIDE + 48] = x3;
            const int bo = boff[v];
            ATOMIC_FMAX_OFF(bo, x0, 0);
            ATOMIC_FMAX_OFF(bo, x1, 64);
            ATOMIC_FMAX_OFF(bo, x2, 128);
            ATOMIC_FMAX_OFF(bo, x3, 192);
        }

        // wave-local transpose barrier: DS ops are in-order per wave
        asm volatile("s_wait_dscnt 0" ::: "memory");

        // coalesced write-out: 16 consecutive lanes emit one 256B row burst (NT)
        float* orow = out + (row0 + half) * 128 + nidx * 4;
#pragma unroll
        for (int j = 0; j < 8; ++j) {
            v4f vv = *(const v4f*)(lds_rd + j * 2 * LDS_STRIDE);
            __builtin_nontemporal_store(vv, (v4f*)(orow + (long)j * 2 * 128));
        }
    }
}

// Remainder rows (N % 16) — scalar path; only launched if a tail exists.
__global__ void tail_rows(const float* __restrict__ feat, const float* __restrict__ W,
                          const float* __restrict__ bias, const int* __restrict__ idx,
                          float* __restrict__ out, float* __restrict__ aggp,
                          long rstart, long N) {
    long t = (long)blockIdx.x * blockDim.x + threadIdx.x;
    long nrow = N - rstart;
    if (t >= nrow * 64) return;
    long r = rstart + (t >> 6);
    int  c = (int)(t & 63);
    float acc = bias[c];
    for (int k = 0; k < 64; ++k) acc += feat[r * 64 + k] * W[c * 64 + k];
    acc = fmaxf(acc, NEG_SLOPE * acc);
    out[r * 128 + c] = acc;
    const int bo = (idx[r] << 8) + c * 4;
    ATOMIC_FMAX_OFF(bo, acc, 0);
}

__global__ void finalize_agg(float* __restrict__ agg, long n) {
    long i = (long)blockIdx.x * blockDim.x + threadIdx.x;
    if (i < n) {
        float v = agg[i];
        // empty segments stayed -inf; mask non-finite to 0 per reference
        bool ok = ((__float_as_uint(v) & 0x7F800000u) != 0x7F800000u);
        agg[i] = ok ? v : 0.0f;
    }
}

__global__ void gather_kernel(const int* __restrict__ idx,
                              const float* __restrict__ agg,
                              float* __restrict__ out, long nq) {
    long t = (long)blockIdx.x * blockDim.x + threadIdx.x;
    if (t >= nq) return;
    const long i = t >> 4;          // point index
    const int  j = (int)(t & 15);   // float4 slot
    const int  s = idx[i];
    v4f v = ((const v4f*)(agg + (size_t)s * 64))[j];      // L2-resident table
    __builtin_nontemporal_store(v, (v4f*)(out + i * 128 + 64) + j);
}

extern "C" void kernel_launch(void* const* d_in, const int* in_sizes, int n_in,
                              void* d_out, int out_size, void* d_ws, size_t ws_size,
                              hipStream_t stream) {
    (void)d_ws; (void)ws_size; (void)n_in;
    const float* feat = (const float*)d_in[0];   // [N, 64]
    const float* W    = (const float*)d_in[1];   // [64, 64]
    const float* bvec = (const float*)d_in[2];   // [64]
    const int*   idx  = (const int*)d_in[3];     // [N]

    float* out = (float*)d_out;                  // [N,128] then [NSEG,64], flat
    const long N        = (long)in_sizes[3];
    const long aggElems = (long)out_size - N * 128;   // NSEG * 64
    float* agg = out + N * 128;

    // 1) init agg table to -inf
    {
        const int th = 256;
        const long bl = (aggElems + th - 1) / th;
        init_agg<<<dim3((unsigned)bl), dim3(th), 0, stream>>>(agg, aggElems);
    }
    // 2) fused WMMA GEMM + bias + leaky_relu + h store + native f32 scatter-max
    {
        const int full_tiles = (int)(N / 16);
        int blocks = 2048;                            // 16,384 waves
        if (blocks * 8 > full_tiles) blocks = (full_tiles + 7) / 8;
        if (full_tiles > 0)
            fused_gemm_scatter<<<dim3(blocks), dim3(256), 0, stream>>>(
                feat, W, bvec, idx, out, agg, full_tiles);
        const long tail = N - (long)full_tiles * 16;
        if (tail > 0) {
            const long work = tail * 64;
            tail_rows<<<dim3((unsigned)((work + 255) / 256)), dim3(256), 0, stream>>>(
                feat, W, bvec, idx, out, agg, (long)full_tiles * 16, N);
        }
    }
    // 3) empty segments -inf -> 0 (in place, agg is output #2)
    {
        const int th = 256;
        const long bl = (aggElems + th - 1) / th;
        finalize_agg<<<dim3((unsigned)bl), dim3(th), 0, stream>>>(agg, aggElems);
    }
    // 4) gather agg[idx] into out[:, 64:128]
    {
        const long nq = N * 16;                       // one float4 per thread
        const int th = 256;
        const long bl = (nq + th - 1) / th;
        gather_kernel<<<dim3((unsigned)bl), dim3(th), 0, stream>>>(idx, agg, out, nq);
    }
}
